// Attention_3736621547687
// MI455X (gfx1250) — matrix operand order repaired and checked by reading, not compiled, not run on hardware
//
#include <hip/hip_runtime.h>
#include <hip/hip_bf16.h>
#include <stdint.h>

// ---------- CDNA5 WMMA types ----------
typedef __attribute__((ext_vector_type(16))) __bf16 v16bf;
typedef __attribute__((ext_vector_type(8)))  float  v8f;

union frag { v16bf v; uint32_t u[8]; };

// Native bf16 conversion (gfx1250 hardware cvt)
__device__ __forceinline__ uint16_t f2bf(float f) {
  union { __bf16 h; uint16_t u; } c; c.h = (__bf16)f; return c.u;
}
__device__ __forceinline__ uint32_t pack2bf(float a, float b) {
  union { __bf16 h[2]; uint32_t u; } c; c.h[0] = (__bf16)a; c.h[1] = (__bf16)b; return c.u;
}

// LDS byte address (wave-relative) for async-to-LDS instructions
__device__ __forceinline__ uint32_t lds_addr(void* p) {
  return (uint32_t)(uintptr_t)(__attribute__((address_space(3))) void*)p;
}
// CDNA5 async DMA: global -> LDS, 16B per lane, tracked by ASYNCcnt
__device__ __forceinline__ void async_copy_b128(uint32_t lds_off, const void* gptr) {
  asm volatile("global_load_async_to_lds_b128 %0, %1, off"
               :: "v"(lds_off), "v"((uint64_t)(uintptr_t)gptr) : "memory");
}
// ASYNCcnt completes in order: wait until <= N outstanding
template <int N>
__device__ __forceinline__ void wait_async_le() {
  asm volatile("s_wait_asynccnt %0" :: "i"(N) : "memory");
}

// =====================================================================
// fp32 -> bf16 streaming convert (one pass per operand; native cvt ops)
// =====================================================================
__global__ void __launch_bounds__(256)
cvt_f32_bf16(const float* __restrict__ src, uint16_t* __restrict__ dst, int n4) {
  int i = blockIdx.x * 256 + threadIdx.x;
  if (i < n4) {
    float4 f = ((const float4*)src)[i];
    uint2 o; o.x = pack2bf(f.x, f.y); o.y = pack2bf(f.z, f.w);
    ((uint2*)dst)[i] = o;
  }
}

// =====================================================================
// GEMM: C[M,N] = A[M,K] @ W[N,K]^T + bias[N]; A,W bf16, C fp32.
// Block tile 64(M) x 128(N), BK=32. 8 waves as 2(M) x 4(N); each wave
// owns a 32x32 region = 2x2 WMMA tiles. Tiles double-buffered via
// async-to-LDS DMA so the next tile streams in during the WMMAs.
// Requires M%64==0, N%128==0, K%32==0 (true for all calls here).
// =====================================================================
#define BM 64
#define BN 128
#define BK 32

__global__ void __launch_bounds__(256)
gemm_bf16wmma(const uint16_t* __restrict__ A, const uint16_t* __restrict__ W,
              const float* __restrict__ bias, float* __restrict__ C,
              int M, int N, int K)
{
  __shared__ uint16_t As[2][BM][BK];   // 2 x 4 KB
  __shared__ uint16_t Ws[2][BN][BK];   // 2 x 8 KB

  const int tid  = threadIdx.x;
  const int lane = tid & 31;
  const int wave = tid >> 5;
  const int wm   = (wave >> 2) * 32;   // 0 or 32
  const int wn   = (wave & 3)  * 32;   // 0,32,64,96
  const int bm   = blockIdx.x * BM;
  const int bn   = blockIdx.y * BN;
  const int half = lane >> 4;
  const int l16  = lane & 15;

  v8f acc[2][2];
  #pragma unroll
  for (int i = 0; i < 2; ++i)
    #pragma unroll
    for (int j = 0; j < 2; ++j) acc[i][j] = (v8f)(0.0f);

  // per-thread staging coordinates (16B = 8 bf16 chunks)
  const int ar = tid >> 2, ac = (tid & 3) * 8;              // A: 256 chunks, 1/thread
  const int wr0 = tid >> 2,          wc0 = (tid & 3) * 8;   // W: 512 chunks, 2/thread
  const int wr1 = (tid + 256) >> 2,  wc1 = wc0;

  // issue 3 async copies for one K-tile into buffer `buf`
  auto stage = [&](int buf, int k0) {
    async_copy_b128(lds_addr(&As[buf][ar][ac]),   A + (size_t)(bm + ar)  * K + k0 + ac);
    async_copy_b128(lds_addr(&Ws[buf][wr0][wc0]), W + (size_t)(bn + wr0) * K + k0 + wc0);
    async_copy_b128(lds_addr(&Ws[buf][wr1][wc1]), W + (size_t)(bn + wr1) * K + k0 + wc1);
  };

  stage(0, 0);                         // prologue
  int buf = 0;

  for (int k0 = 0; k0 < K; k0 += BK) {
    const bool has_next = (k0 + BK < K);
    if (has_next) stage(buf ^ 1, k0 + BK);   // overlap next DMA with this compute
    if (has_next) wait_async_le<3>();        // 3 oldest = current tile done
    else          wait_async_le<0>();
    __syncthreads();

    // ---- fragments from LDS, contiguous k order (k = 16*half + s) for A and B ----
    frag afr[2], bfr[2];
    #pragma unroll
    for (int mi = 0; mi < 2; ++mi) {
      const uint16_t* ap = &As[buf][wm + mi * 16 + l16][0];
      const int kbase = half * 16;           // lanes 0-15: K 0..15; 16-31: 16..31
      #pragma unroll
      for (int i = 0; i < 8; ++i) {
        int k = kbase + 2 * i;
        afr[mi].u[i] = *(const uint32_t*)(ap + k);
      }
    }
    #pragma unroll
    for (int ni = 0; ni < 2; ++ni) {
      const uint16_t* bp = &Ws[buf][wn + ni * 16 + l16][0];
      const int kbase = half * 16;           // lanes 0-15: K 0..15; 16-31: 16..31
      #pragma unroll
      for (int i = 0; i < 8; ++i)
        bfr[ni].u[i] = *(const uint32_t*)(bp + kbase + 2 * i);
    }
    #pragma unroll
    for (int mi = 0; mi < 2; ++mi)
      #pragma unroll
      for (int ni = 0; ni < 2; ++ni)
        acc[mi][ni] = __builtin_amdgcn_wmma_f32_16x16x32_bf16(
            false, afr[mi].v, false, bfr[ni].v, (short)0, acc[mi][ni], false, false);
    __syncthreads();
    buf ^= 1;
  }

  // ---- epilogue: C layout lane<16: M=r, N=lane; lane>=16: M=r+8 ----
  #pragma unroll
  for (int mi = 0; mi < 2; ++mi)
    #pragma unroll
    for (int ni = 0; ni < 2; ++ni) {
      int n  = bn + wn + ni * 16 + l16;
      int m0 = bm + wm + mi * 16 + half * 8;
      float b = bias[n];
      #pragma unroll
      for (int r = 0; r < 8; ++r)
        C[(size_t)(m0 + r) * N + n] = acc[mi][ni][r] + b;
    }
}

// =====================================================================
// RMSNorm + RoPE + scale + fp32->bf16. One wave per head per token.
// qkv row: [16 q heads | 4 k heads | 4 v heads] x 128
// =====================================================================
__global__ void __launch_bounds__(256)
qkv_post(const float* __restrict__ qkv, const float* __restrict__ cosb,
         const float* __restrict__ sinb, const float* __restrict__ qw,
         const float* __restrict__ kw, uint16_t* __restrict__ qb,
         uint16_t* __restrict__ kb, uint16_t* __restrict__ vb)
{
  const int s    = blockIdx.x;
  const int lane = threadIdx.x & 31;
  const int wave = threadIdx.x >> 5;
  const float* row = qkv + (size_t)s * 3072;
  const float* cs  = cosb + s * 64;
  const float* sn  = sinb + s * 64;

  for (int hh = wave; hh < 24; hh += 8) {
    const float* x; uint16_t* outp; int kind;
    if (hh < 16)      { x = row + hh * 128;                 outp = qb + ((size_t)s * 16 + hh) * 128;        kind = 0; }
    else if (hh < 20) { x = row + 2048 + (hh - 16) * 128;   outp = kb + ((size_t)s * 4 + (hh - 16)) * 128;  kind = 1; }
    else              { x = row + 2560 + (hh - 20) * 128;   outp = vb + ((size_t)s * 4 + (hh - 20)) * 128;  kind = 2; }

    float x0 = x[lane], x1 = x[lane + 32], x2 = x[lane + 64], x3 = x[lane + 96];
    if (kind == 2) {               // V: passthrough convert
      outp[lane]      = f2bf(x0); outp[lane + 32] = f2bf(x1);
      outp[lane + 64] = f2bf(x2); outp[lane + 96] = f2bf(x3);
      continue;
    }
    float ss = x0 * x0 + x1 * x1 + x2 * x2 + x3 * x3;
    #pragma unroll
    for (int m = 16; m >= 1; m >>= 1) ss += __shfl_xor(ss, m, 32);
    float rs = rsqrtf(ss * (1.0f / 128.0f) + 1e-6f);
    const float* w = (kind == 0) ? qw : kw;
    x0 *= rs * w[lane]; x1 *= rs * w[lane + 32];
    x2 *= rs * w[lane + 64]; x3 *= rs * w[lane + 96];
    // RoPE pairs (d, d+64): this lane owns d=lane and d=lane+32
    float c0 = cs[lane], s0 = sn[lane], c1 = cs[lane + 32], s1 = sn[lane + 32];
    float o0 = x0 * c0 - x2 * s0;
    float o2 = x0 * s0 + x2 * c0;
    float o1 = x1 * c1 - x3 * s1;
    float o3 = x1 * s1 + x3 * c1;
    const float sc = (kind == 0) ? 0.08838834764831845f : 1.0f;  // 128^-0.5 folded into Q
    outp[lane]      = f2bf(o0 * sc); outp[lane + 32] = f2bf(o1 * sc);
    outp[lane + 64] = f2bf(o2 * sc); outp[lane + 96] = f2bf(o3 * sc);
  }
}

// =====================================================================
// Flash attention (causal, GQA). Block = 4 waves (one per group g of the
// kv head), each wave owns one 16-query tile of one head. 32 keys/step:
// scores via 4 WMMAs (K=128), online softmax with half-wave row
// reductions, P transposed C-layout -> A-layout through LDS, PV via 8
// WMMAs. Output written directly as bf16 (feeds the second GEMM).
// =====================================================================
__global__ void __launch_bounds__(128)
flash_attn(const uint16_t* __restrict__ qb, const uint16_t* __restrict__ kb,
           const uint16_t* __restrict__ vb, uint16_t* __restrict__ O)
{
  __shared__ uint16_t Pld[4][16][32];    // per-wave P staging (bf16)

  const int qt   = blockIdx.x;           // query tile, 0..127
  const int kvh  = blockIdx.y;           // kv head, 0..3
  const int g    = threadIdx.x >> 5;     // group within kv head
  const int lane = threadIdx.x & 31;
  const int h    = kvh * 4 + g;          // full head index
  const int half = lane >> 4, l16 = lane & 15;
  const int qbase = qt * 16;

  // Q fragments for 4 K=32 dim chunks (A-matrix layout)
  frag qf[4];
  {
    const uint16_t* qp = qb + ((size_t)(qbase + l16) * 16 + h) * 128;
    const int kbase = half * 16;
    #pragma unroll
    for (int c = 0; c < 4; ++c)
      #pragma unroll
      for (int i = 0; i < 8; ++i) {
        int k = c * 32 + kbase + 2 * i;
        qf[c].u[i] = *(const uint32_t*)(qp + k);
      }
  }

  v8f o_acc[8];
  #pragma unroll
  for (int i = 0; i < 8; ++i) o_acc[i] = (v8f)(0.0f);
  float mrow[8], lrow[8];
  #pragma unroll
  for (int r = 0; r < 8; ++r) { mrow[r] = -1e30f; lrow[r] = 0.0f; }

  const int nsteps = (qbase + 16 + 31) >> 5;   // causal: keys 0..qbase+15

  for (int st = 0; st < nsteps; ++st) {
    const int key0 = st * 32;

    // ---- scores: two 16-key subtiles ----
    v8f sacc[2];
    #pragma unroll
    for (int sub = 0; sub < 2; ++sub) {
      sacc[sub] = (v8f)(0.0f);
      const int kcol = key0 + sub * 16 + l16;          // this lane's B column (key)
      const uint16_t* kp = kb + ((size_t)kcol * 4 + kvh) * 128;
      #pragma unroll
      for (int c = 0; c < 4; ++c) {
        frag bf;
        const int kbase = c * 32 + half * 16;
        #pragma unroll
        for (int i = 0; i < 8; ++i)
          bf.u[i] = *(const uint32_t*)(kp + kbase + 2 * i);
        sacc[sub] = __builtin_amdgcn_wmma_f32_16x16x32_bf16(
            false, qf[c].v, false, bf.v, (short)0, sacc[sub], false, false);
      }
      #pragma unroll
      for (int r = 0; r < 8; ++r) {                    // causal mask
        int qrow = qbase + half * 8 + r;
        if (kcol > qrow) sacc[sub][r] = -1e9f;
      }
    }

    // ---- online softmax (row stats reduced across the 16-lane half) ----
    float pscale[8];
    #pragma unroll
    for (int r = 0; r < 8; ++r) {
      float mloc = fmaxf(sacc[0][r], sacc[1][r]);
      #pragma unroll
      for (int msk = 8; msk >= 1; msk >>= 1) mloc = fmaxf(mloc, __shfl_xor(mloc, msk, 32));
      float mnew = fmaxf(mrow[r], mloc);
      float corr = __expf(mrow[r] - mnew);
      float p0 = __expf(sacc[0][r] - mnew);
      float p1 = __expf(sacc[1][r] - mnew);
      float ploc = p0 + p1;
      #pragma unroll
      for (int msk = 8; msk >= 1; msk >>= 1) ploc += __shfl_xor(ploc, msk, 32);
      lrow[r] = lrow[r] * corr + ploc;
      mrow[r] = mnew;
      sacc[0][r] = p0; sacc[1][r] = p1;
      pscale[r] = corr;
    }
    #pragma unroll
    for (int c = 0; c < 8; ++c)
      #pragma unroll
      for (int r = 0; r < 8; ++r) o_acc[c][r] *= pscale[r];

    // ---- transpose P (C layout) -> A layout via LDS ----
    #pragma unroll
    for (int sub = 0; sub < 2; ++sub)
      #pragma unroll
      for (int r = 0; r < 8; ++r)
        Pld[g][half * 8 + r][sub * 16 + l16] = f2bf(sacc[sub][r]);
    __syncthreads();

    frag pf;
    {
      const uint16_t* pp = &Pld[g][l16][0];
      const int kbase = half * 16;
      #pragma unroll
      for (int i = 0; i < 8; ++i) {
        int k = kbase + 2 * i;
        pf.u[i] = *(const uint32_t*)(pp + k);
      }
    }

    // ---- PV: 8 chunks of 16 dims; B[k=key][n=dim] gathered from V ----
    #pragma unroll
    for (int c = 0; c < 8; ++c) {
      frag vf;
      const int keyb = key0 + half * 16;
      const int dim  = c * 16 + l16;
      #pragma unroll
      for (int i = 0; i < 8; ++i) {
        int kk = keyb + 2 * i;
        uint32_t lo = vb[((size_t)kk * 4 + kvh) * 128 + dim];
        uint32_t hi = vb[((size_t)(kk + 1) * 4 + kvh) * 128 + dim];
        vf.u[i] = lo | (hi << 16);
      }
      o_acc[c] = __builtin_amdgcn_wmma_f32_16x16x32_bf16(
          false, pf.v, false, vf.v, (short)0, o_acc[c], false, false);
    }
    __syncthreads();
  }

  // ---- epilogue: normalize and store bf16 (feeds GEMM2 directly) ----
  #pragma unroll
  for (int r = 0; r < 8; ++r) {
    float inv = 1.0f / lrow[r];
    int qrow = qbase + half * 8 + r;
    #pragma unroll
    for (int c = 0; c < 8; ++c)
      O[(size_t)qrow * 2048 + h * 128 + c * 16 + l16] = f2bf(o_acc[c][r] * inv);
  }
}

// =====================================================================
extern "C" void kernel_launch(void* const* d_in, const int* in_sizes, int n_in,
                              void* d_out, int out_size, void* d_ws, size_t ws_size,
                              hipStream_t stream) {
  const float* hidden = (const float*)d_in[0];
  const float* cosb   = (const float*)d_in[1];
  const float* sinb   = (const float*)d_in[2];
  // d_in[3] k_cache, d_in[4] v_cache, d_in[5] mask, d_in[12] kv_write_indices: unused
  const float* wqkv   = (const float*)d_in[6];
  const float* bqkv   = (const float*)d_in[7];
  const float* wo     = (const float*)d_in[8];
  const float* bo     = (const float*)d_in[9];
  const float* qw     = (const float*)d_in[10];
  const float* kw     = (const float*)d_in[11];
  float* out = (float*)d_out;

  char* ws = (char*)d_ws;
  uint16_t* hid_bf  = (uint16_t*)ws;  ws += (size_t)2048 * 2048 * 2;
  uint16_t* wqkv_bf = (uint16_t*)ws;  ws += (size_t)3072 * 2048 * 2;
  uint16_t* wo_bf   = (uint16_t*)ws;  ws += (size_t)2048 * 2048 * 2;
  float*    qkv     = (float*)ws;     ws += (size_t)2048 * 3072 * 4;
  uint16_t* qb      = (uint16_t*)ws;  ws += (size_t)2048 * 16 * 128 * 2;
  uint16_t* kbuf    = (uint16_t*)ws;  ws += (size_t)2048 * 4 * 128 * 2;
  uint16_t* vbuf    = (uint16_t*)ws;  ws += (size_t)2048 * 4 * 128 * 2;
  uint16_t* attn_bf = (uint16_t*)ws;  ws += (size_t)2048 * 2048 * 2;

  // 0) one-time fp32 -> bf16 conversions (streaming, native cvt ops)
  cvt_f32_bf16<<<(2048 * 2048 / 4 + 255) / 256, 256, 0, stream>>>(hidden, hid_bf,  2048 * 2048 / 4);
  cvt_f32_bf16<<<(3072 * 2048 / 4 + 255) / 256, 256, 0, stream>>>(wqkv,   wqkv_bf, 3072 * 2048 / 4);
  cvt_f32_bf16<<<(2048 * 2048 / 4 + 255) / 256, 256, 0, stream>>>(wo,     wo_bf,   2048 * 2048 / 4);

  // 1) qkv = hidden @ wqkv^T + bqkv   (M=2048, N=3072, K=2048)
  gemm_bf16wmma<<<dim3(2048 / BM, 3072 / BN), 256, 0, stream>>>(
      hid_bf, wqkv_bf, bqkv, qkv, 2048, 3072, 2048);
  // 2) RMSNorm + RoPE + scale -> bf16 q/k/v
  qkv_post<<<2048, 256, 0, stream>>>(qkv, cosb, sinb, qw, kw, qb, kbuf, vbuf);
  // 3) causal GQA flash attention -> attn bf16 [2048, 2048]
  flash_attn<<<dim3(128, 4), 128, 0, stream>>>(qb, kbuf, vbuf, attn_bf);
  // 4) out = attn @ wo^T + bo   (M=2048, N=2048, K=2048, fp32 out)
  gemm_bf16wmma<<<dim3(2048 / BM, 2048 / BN), 256, 0, stream>>>(
      attn_bf, wo_bf, bo, out, 2048, 2048, 2048);
}
